// MultiheadTransformer_71829033058508
// MI455X (gfx1250) — matrix operand order repaired and checked by reading, not compiled, not run on hardware
//
#include <hip/hip_runtime.h>
#include <hip/hip_bf16.h>

// ---------------- problem constants ----------------
#define NN   100000
#define EE   1600000
#define CCH  128        // in/out channels
#define HDIM 128        // H*D
#define NHEAD 4
#define DHEAD 32

typedef __attribute__((ext_vector_type(16))) _Float16 v16h;
typedef __attribute__((ext_vector_type(8)))  _Float16 v8h;
typedef __attribute__((ext_vector_type(4)))  _Float16 v4h;
typedef __attribute__((ext_vector_type(8)))  float    v8f;

#define WMMA_F16(a, b, c) \
    __builtin_amdgcn_wmma_f32_16x16x32_f16(false, (a), false, (b), (short)0, (c), false, false)

// order-preserving float<->uint encoding for atomicMax-based segment max
__device__ __forceinline__ unsigned enc_f(float f) {
    unsigned u = __float_as_uint(f);
    return (u & 0x80000000u) ? ~u : (u | 0x80000000u);
}
__device__ __forceinline__ float dec_f(unsigned u) {
    return __uint_as_float((u & 0x80000000u) ? (u & 0x7FFFFFFFu) : ~u);
}

// ---------------- init workspace ----------------
__global__ __launch_bounds__(256) void init_ws_k(float* sums, unsigned* amax, float* denom) {
    int i = blockIdx.x * 256 + threadIdx.x;
    if (i < 256) sums[i] = 0.f;
    if (i < NN * NHEAD) { amax[i] = 0u; denom[i] = 0.f; }
}

// ---------------- weights: f32 -> f16, transposed to [outchan][k] ----------------
__global__ __launch_bounds__(256) void weight_prep_k(
    const float* __restrict__ Wq, const float* __restrict__ Wk,
    const float* __restrict__ Wv, const float* __restrict__ Ws,
    const float* __restrict__ Wfc,
    _Float16* __restrict__ WqkvsT,  // [512][128]  (outchan-major)
    _Float16* __restrict__ WfcT)    // [128][128]
{
    int i = blockIdx.x * 256 + threadIdx.x;
    if (i < CCH * 512) {
        int n = i >> 7, k = i & 127;              // n: region*128+col
        int region = n >> 7, col = n & 127;
        const float* W = (region == 0) ? Wq : (region == 1) ? Wk : (region == 2) ? Wv : Ws;
        WqkvsT[i] = (_Float16)W[k * HDIM + col];  // W is [C=128, HD=128] row-major
    } else if (i < CCH * 512 + CCH * CCH) {
        int j = i - CCH * 512;
        int n = j >> 7, k = j & 127;
        WfcT[j] = (_Float16)Wfc[k * CCH + n];     // Wfc is [HD=128, C=128]
    }
}

// ---------------- BatchNorm statistics ----------------
__global__ __launch_bounds__(256) void bn_stats_k(const float* __restrict__ x,
                                                  float* __restrict__ sums) {
    __shared__ float s0[256], s1[256];
    int c  = threadIdx.x & 127;
    int rg = threadIdx.x >> 7;   // two row-groups per block
    float s = 0.f, sq = 0.f;
    for (int r = blockIdx.x * 2 + rg; r < NN; r += gridDim.x * 2) {
        float v = x[(size_t)r * CCH + c];
        s += v; sq += v * v;
    }
    s0[threadIdx.x] = s; s1[threadIdx.x] = sq;
    __syncthreads();
    if (rg == 0) {
        atomicAdd(&sums[c],       s0[threadIdx.x] + s0[threadIdx.x + 128]);
        atomicAdd(&sums[128 + c], s1[threadIdx.x] + s1[threadIdx.x + 128]);
    }
}

// ---------------- BN apply + ReLU, emit f16 activations ----------------
__global__ __launch_bounds__(256) void bn_apply_k(
    const float* __restrict__ x, const float* __restrict__ sums,
    const float* __restrict__ gamma, const float* __restrict__ beta,
    _Float16* __restrict__ h16)
{
    size_t i = (size_t)blockIdx.x * 256 + threadIdx.x;
    if (i >= (size_t)NN * CCH) return;
    int c = (int)(i & 127);
    float inv = 1.0f / (float)NN;
    float mu  = sums[c] * inv;
    float var = sums[128 + c] * inv - mu * mu;
    float rs  = rsqrtf(var + 1e-5f);
    float v   = (x[i] - mu) * rs * gamma[c] + beta[c];
    h16[i] = (_Float16)fmaxf(v, 0.f);
}

// ---------------- fused Q/K/V/Skip GEMM via WMMA ----------------
// Per block: 32 rows x 512 out-cols. 8 waves; wave owns ONE region
// (region = wave>>1: q/k/v/skip) and one 16-row tile (rt = wave&1),
// and sweeps that region's 8 column tiles with double-buffered B frags.
__global__ __launch_bounds__(256) void qkvs_gemm_k(
    const _Float16* __restrict__ h16, const _Float16* __restrict__ WqkvsT,
    const float* __restrict__ bq, const float* __restrict__ bk,
    const float* __restrict__ bv, const float* __restrict__ bskip,
    _Float16* __restrict__ q16, _Float16* __restrict__ k16,
    _Float16* __restrict__ v16, float* __restrict__ agg)
{
    __shared__ _Float16 hA[32][136];               // +8 halves pad (16B) vs bank conflicts
    const int rowbase = blockIdx.x * 32;
    const int tid = threadIdx.x;
    {   // stage 32x128 activation tile (coalesced 32B per lane)
        int r  = tid >> 3;
        int c0 = (tid & 7) * 16;
        const _Float16* src = h16 + (size_t)(rowbase + r) * CCH + c0;
        *(v8h*)(&hA[r][c0])     = *(const v8h*)(src);
        *(v8h*)(&hA[r][c0 + 8]) = *(const v8h*)(src + 8);
    }
    __syncthreads();

    const int wave   = tid >> 5, lane = tid & 31;
    const int region = wave >> 1;                  // 0:q 1:k 2:v 3:skip
    const int rt     = wave & 1;
    const int lrow   = lane & 15;
    const int hi     = lane >> 4;

    // A fragments (contiguous k order, same as B: lanes 0-15 hold K k0..k0+15,
    // lanes 16-31 shifted by 16) — one fragment per K-block, reused over 8 col tiles.
    v16h afrag[4];
    {
        int arow = rt * 16 + lrow;
        #pragma unroll
        for (int kb = 0; kb < 4; ++kb) {
            int k0 = kb * 32 + hi * 16;
            union { v16h v; v8h h[2]; } u;
            u.h[0] = *(const v8h*)(&hA[arow][k0]);
            u.h[1] = *(const v8h*)(&hA[arow][k0 + 8]);
            afrag[kb] = u.v;
        }
    }

    // this wave's 128-col weight slice (B: lanes 0-15 K 0..15, lanes 16-31 K 16..31)
    const _Float16* wbase = WqkvsT + (size_t)region * CCH * CCH + (size_t)lrow * CCH + hi * 16;
    const int mbase = rowbase + rt * 16 + hi * 8;  // C/D layout: VGPR j -> M=j (+8 hi half)

    v16h bcur[4], bnxt[4];
    #pragma unroll
    for (int kb = 0; kb < 4; ++kb)
        bcur[kb] = *(const v16h*)(wbase + kb * 32);

    if (region == 3) {                             // skip -> f32 agg (initializes agg)
        #pragma unroll
        for (int i = 0; i < 8; ++i) {
            if (i < 7) {                           // prefetch next tile's B frags
                #pragma unroll
                for (int kb = 0; kb < 4; ++kb)
                    bnxt[kb] = *(const v16h*)(wbase + (size_t)(i + 1) * 16 * CCH + kb * 32);
            }
            v8f acc = {};
            #pragma unroll
            for (int kb = 0; kb < 4; ++kb) acc = WMMA_F16(afrag[kb], bcur[kb], acc);
            int   ocol = i * 16 + lrow;
            float b    = bskip[ocol];
            #pragma unroll
            for (int j = 0; j < 8; ++j)
                agg[(size_t)(mbase + j) * HDIM + ocol] = acc[j] + b;
            #pragma unroll
            for (int kb = 0; kb < 4; ++kb) bcur[kb] = bnxt[kb];
        }
    } else {                                       // q/k/v -> f16 (uniform per wave)
        const float* bias = (region == 0) ? bq : (region == 1) ? bk : bv;
        _Float16*    dstp = (region == 0) ? q16 : (region == 1) ? k16 : v16;
        #pragma unroll
        for (int i = 0; i < 8; ++i) {
            if (i < 7) {
                #pragma unroll
                for (int kb = 0; kb < 4; ++kb)
                    bnxt[kb] = *(const v16h*)(wbase + (size_t)(i + 1) * 16 * CCH + kb * 32);
            }
            v8f acc = {};
            #pragma unroll
            for (int kb = 0; kb < 4; ++kb) acc = WMMA_F16(afrag[kb], bcur[kb], acc);
            int   ocol = i * 16 + lrow;
            float b    = bias[ocol];
            #pragma unroll
            for (int j = 0; j < 8; ++j)
                dstp[(size_t)(mbase + j) * HDIM + ocol] = (_Float16)(acc[j] + b);
            #pragma unroll
            for (int kb = 0; kb < 4; ++kb) bcur[kb] = bnxt[kb];
        }
    }
}

// ---------------- edge phase 1: alpha + segment max ----------------
// one wave32 per edge; lane -> head (lane>>3), 4 halves per lane.
__global__ __launch_bounds__(256) void edge_alpha_k(
    const int* __restrict__ ei, const _Float16* __restrict__ q16,
    const _Float16* __restrict__ k16, float* __restrict__ alpha,
    unsigned* __restrict__ amax)
{
    int e = blockIdx.x * 8 + (threadIdx.x >> 5);
    if (e >= EE) return;
    int lane = threadIdx.x & 31;
    int src = ei[e];
    int dst = ei[EE + e];
    int h   = lane >> 3;
    int off = h * DHEAD + (lane & 7) * 4;
    v4h qv = *(const v4h*)(q16 + (size_t)dst * HDIM + off);
    v4h kv = *(const v4h*)(k16 + (size_t)src * HDIM + off);
    float s = 0.f;
    #pragma unroll
    for (int j = 0; j < 4; ++j) s += (float)qv[j] * (float)kv[j];
    s += __shfl_xor(s, 1, 32);
    s += __shfl_xor(s, 2, 32);
    s += __shfl_xor(s, 4, 32);
    if ((lane & 7) == 0) {
        float a = s * 0.17677669529663687f;        // 1/sqrt(32)
        alpha[(size_t)e * NHEAD + h] = a;
        atomicMax(&amax[(size_t)dst * NHEAD + h], enc_f(a));
    }
}

// ---------------- edge phase 2: exp + segment sum ----------------
__global__ __launch_bounds__(256) void edge_ex_k(
    const int* __restrict__ ei, float* __restrict__ alpha,
    const unsigned* __restrict__ amax, float* __restrict__ denom)
{
    size_t i = (size_t)blockIdx.x * 256 + threadIdx.x;
    if (i >= (size_t)EE * NHEAD) return;
    int e = (int)(i >> 2), h = (int)(i & 3);
    int dst = ei[EE + e];
    float m  = dec_f(amax[(size_t)dst * NHEAD + h]);
    float ex = __expf(alpha[i] - m);
    alpha[i] = ex;
    atomicAdd(&denom[(size_t)dst * NHEAD + h], ex);
}

// ---------------- edge phase 3: weighted scatter-add of V ----------------
__global__ __launch_bounds__(256) void edge_agg_k(
    const int* __restrict__ ei, const float* __restrict__ exv,
    const float* __restrict__ denom, const _Float16* __restrict__ v16,
    float* __restrict__ agg)
{
    int e = blockIdx.x * 8 + (threadIdx.x >> 5);
    if (e >= EE) return;
    int lane = threadIdx.x & 31;
    int src = ei[e], dst = ei[EE + e];
    int h   = lane >> 3;
    int off = h * DHEAD + (lane & 7) * 4;
    float w = exv[(size_t)e * NHEAD + h] / denom[(size_t)dst * NHEAD + h];
    v4h vv = *(const v4h*)(v16 + (size_t)src * HDIM + off);
    float* ap = agg + (size_t)dst * HDIM + off;
    #pragma unroll
    for (int j = 0; j < 4; ++j) atomicAdd(ap + j, w * (float)vv[j]);
}

// ---------------- final GEMM: out = conv @ Wfc + bfc + x ----------------
__global__ __launch_bounds__(256) void out_gemm_k(
    const float* __restrict__ agg, const _Float16* __restrict__ WfcT,
    const float* __restrict__ bfc, const float* __restrict__ xres,
    float* __restrict__ out)
{
    __shared__ _Float16 cA[32][136];
    const int rowbase = blockIdx.x * 32;
    const int tid = threadIdx.x;
    {   // stage + f32->f16 convert conv tile (agg holds attn-agg + skip + bias)
        int r  = tid >> 3;
        int c0 = (tid & 7) * 16;
        const float* src = agg + (size_t)(rowbase + r) * HDIM + c0;
        #pragma unroll
        for (int j = 0; j < 16; ++j) cA[r][c0 + j] = (_Float16)src[j];
    }
    __syncthreads();

    const int wave = tid >> 5, lane = tid & 31;
    const int rt   = wave & 1;
    const int lrow = lane & 15;
    const int hi   = lane >> 4;

    v16h afrag[4];
    {
        int arow = rt * 16 + lrow;
        #pragma unroll
        for (int kb = 0; kb < 4; ++kb) {
            int k0 = kb * 32 + hi * 16;
            union { v16h v; v8h h[2]; } u;
            u.h[0] = *(const v8h*)(&cA[arow][k0]);
            u.h[1] = *(const v8h*)(&cA[arow][k0 + 8]);
            afrag[kb] = u.v;
        }
    }

    const _Float16* wbase = WfcT + (size_t)lrow * CCH + hi * 16;
    const int ct0 = (wave >> 1) * 2;               // 2 col tiles per wave

    // preload both tiles' B fragments, then run back-to-back wmma chains
    v16h b0[4], b1[4];
    #pragma unroll
    for (int kb = 0; kb < 4; ++kb) {
        b0[kb] = *(const v16h*)(wbase + (size_t)(ct0 + 0) * 16 * CCH + kb * 32);
        b1[kb] = *(const v16h*)(wbase + (size_t)(ct0 + 1) * 16 * CCH + kb * 32);
    }

    #pragma unroll
    for (int i = 0; i < 2; ++i) {
        int col = (ct0 + i) * 16 + lrow;           // 0..127
        v8f acc = {};
        #pragma unroll
        for (int kb = 0; kb < 4; ++kb)
            acc = WMMA_F16(afrag[kb], (i == 0) ? b0[kb] : b1[kb], acc);
        int   mbase = rowbase + rt * 16 + hi * 8;
        float b     = bfc[col];
        #pragma unroll
        for (int j = 0; j < 8; ++j) {
            size_t idx = (size_t)(mbase + j) * CCH + col;
            out[idx] = acc[j] + b + xres[idx];     // + residual
        }
    }
}

// ---------------- host: workspace layout + launch sequence ----------------
extern "C" void kernel_launch(void* const* d_in, const int* in_sizes, int n_in,
                              void* d_out, int out_size, void* d_ws, size_t ws_size,
                              hipStream_t stream) {
    (void)in_sizes; (void)n_in; (void)out_size; (void)ws_size;
    const float* x     = (const float*)d_in[0];
    const int*   ei    = (const int*)d_in[1];        // [2,E] int32
    const float* gamma = (const float*)d_in[2];
    const float* beta  = (const float*)d_in[3];
    const float* Wq    = (const float*)d_in[4];
    const float* bq    = (const float*)d_in[5];
    const float* Wk    = (const float*)d_in[6];
    const float* bk    = (const float*)d_in[7];
    const float* Wv    = (const float*)d_in[8];
    const float* bv    = (const float*)d_in[9];
    const float* Wsk   = (const float*)d_in[10];
    const float* bsk   = (const float*)d_in[11];
    const float* Wfc   = (const float*)d_in[12];
    const float* bfc   = (const float*)d_in[13];
    float* out = (float*)d_out;

    uint8_t* w = (uint8_t*)d_ws;
    size_t off = 0;
    auto take = [&](size_t bytes) { uint8_t* p = w + off; off += (bytes + 255) & ~(size_t)255; return p; };

    float*     sums   = (float*)take(256 * sizeof(float));            // mean/var partials
    _Float16*  h16    = (_Float16*)take((size_t)NN * CCH * 2);
    _Float16*  q16    = (_Float16*)take((size_t)NN * HDIM * 2);
    _Float16*  k16    = (_Float16*)take((size_t)NN * HDIM * 2);
    _Float16*  v16    = (_Float16*)take((size_t)NN * HDIM * 2);
    float*     agg    = (float*)take((size_t)NN * HDIM * 4);          // skip then += attn agg
    float*     alpha  = (float*)take((size_t)EE * NHEAD * 4);         // alpha, then ex in place
    unsigned*  amax   = (unsigned*)take((size_t)NN * NHEAD * 4);
    float*     denom  = (float*)take((size_t)NN * NHEAD * 4);
    _Float16*  WqkvsT = (_Float16*)take((size_t)512 * CCH * 2);
    _Float16*  WfcT   = (_Float16*)take((size_t)CCH * CCH * 2);

    init_ws_k   <<<(NN * NHEAD + 255) / 256, 256, 0, stream>>>(sums, amax, denom);
    weight_prep_k<<<(CCH * 512 + CCH * CCH + 255) / 256, 256, 0, stream>>>(Wq, Wk, Wv, Wsk, Wfc, WqkvsT, WfcT);
    bn_stats_k  <<<400, 256, 0, stream>>>(x, sums);
    bn_apply_k  <<<(NN * CCH + 255) / 256, 256, 0, stream>>>(x, sums, gamma, beta, h16);
    qkvs_gemm_k <<<NN / 32, 256, 0, stream>>>(h16, WqkvsT, bq, bk, bv, bsk, q16, k16, v16, agg);
    edge_alpha_k<<<EE / 8, 256, 0, stream>>>(ei, q16, k16, alpha, amax);
    edge_ex_k   <<<(EE * NHEAD + 255) / 256, 256, 0, stream>>>(ei, alpha, amax, denom);
    edge_agg_k  <<<EE / 8, 256, 0, stream>>>(ei, alpha, denom, v16, agg);
    out_gemm_k  <<<NN / 32, 256, 0, stream>>>(agg, WfcT, bfc, x, out);
}
